// GNNLayer_20091857010801
// MI455X (gfx1250) — matrix operand order repaired and checked by reading, not compiled, not run on hardware
//
#include <hip/hip_runtime.h>
#include <hip/hip_bf16.h>
#include <math.h>

typedef __attribute__((ext_vector_type(16))) _Float16 v16h;
typedef __attribute__((ext_vector_type(8)))  _Float16 v8h;
typedef __attribute__((ext_vector_type(8)))  float    v8f;

#define FDIM 128
#define GDIM 4096
#define UDIM 100000
#define IDIM 8192
#define MDIM 8192
#define NEG_SLOPE 0.01f
#define ISPLIT 32

// ---------------------------------------------------------------------------
// WMMA operand loaders (CDNA5 wave32, V_WMMA_F32_16X16X32_F16)
// A (16x32, MxK): lane l -> m=l&15, half h=l>>4; slots 0..7 <-> K=8h..8h+7,
//                 slots 8..15 <-> K=16+8h..16+8h+7   (two contiguous 16B runs)
// B (32x16, KxN): lane l -> n=l&15, half h=l>>4; slots 0..7 <-> K=8h..8h+7,
//                 slots 8..15 <-> K=16+8h..16+8h+7 (two 16B runs, fed from B^T row-major)
// ---------------------------------------------------------------------------
static __device__ __forceinline__ v16h load_a_f16(const _Float16* __restrict__ X, int ld,
                                                  int row0, int k0, int lane) {
  const int m  = lane & 15;
  const int kb = (lane >> 4) << 3;
  const _Float16* p = X + (size_t)(row0 + m) * ld + k0;
  v8h lo = *(const v8h*)(p + kb);
  v8h hi = *(const v8h*)(p + 16 + kb);
  v16h a;
#pragma unroll
  for (int i = 0; i < 8; ++i) { a[i] = lo[i]; a[8 + i] = hi[i]; }
  return a;
}

static __device__ __forceinline__ v16h load_b_f16(const _Float16* __restrict__ Bt, int ld,
                                                  int row0, int k0, int lane) {
  const int n  = lane & 15;
  const int kb = (lane >> 4) << 3;
  const _Float16* p = Bt + (size_t)(row0 + n) * ld + k0;
  v8h lo = *(const v8h*)(p + kb);
  v8h hi = *(const v8h*)(p + 16 + kb);
  v16h b;
#pragma unroll
  for (int i = 0; i < 8; ++i) { b[i] = lo[i]; b[8 + i] = hi[i]; }
  return b;
}

// f32 A loader with optional elementwise multiplier (for the lin5 fused inputs)
static __device__ __forceinline__ v16h load_a_f32(const float* __restrict__ X, int ld,
                                                  int row0, int k0,
                                                  const float* __restrict__ Mul, int lane) {
  const int m  = lane & 15;
  const int kb = (lane >> 4) << 3;
  const float* p = X + (size_t)(row0 + m) * ld + k0;
  const float* q = Mul ? (Mul + (size_t)(row0 + m) * ld + k0) : nullptr;
  v16h a;
#pragma unroll
  for (int v = 0; v < 8; ++v) {
    const int k = ((v < 4) ? (2 * v) : (16 + 2 * (v - 4))) + kb;
#pragma unroll
    for (int e = 0; e < 2; ++e) {
      float x = p[k + e];
      if (q) x *= q[k + e];
      a[2 * v + e] = (_Float16)x;
    }
  }
  return a;
}

static __device__ __forceinline__ v16h load_a_lds(const _Float16* __restrict__ X, int ld,
                                                  int k0, int lane) {
  const int m  = lane & 15;
  const int kb = (lane >> 4) << 3;
  const _Float16* p = X + m * ld + k0;
  v8h lo = *(const v8h*)(p + kb);
  v8h hi = *(const v8h*)(p + 16 + kb);
  v16h a;
#pragma unroll
  for (int i = 0; i < 8; ++i) { a[i] = lo[i]; a[8 + i] = hi[i]; }
  return a;
}

static __device__ __forceinline__ v8f wmma16(v16h a, v16h b, v8f c) {
  return __builtin_amdgcn_wmma_f32_16x16x32_f16(false, a, false, b, (short)0, c, false, false);
}

// ---------------------------------------------------------------------------
// Utility kernels
// ---------------------------------------------------------------------------
__global__ void k_zero(float* __restrict__ p, long long n) {
  long long i = (long long)blockIdx.x * blockDim.x + threadIdx.x;
  if (i < n) p[i] = 0.f;
}

__global__ void k_cvt_f16(const float* __restrict__ x, _Float16* __restrict__ y, int n) {
  int i = blockIdx.x * blockDim.x + threadIdx.x;
  if (i < n) y[i] = (_Float16)x[i];
}

__global__ void k_gather_members(const float* __restrict__ user_emb,
                                 const int* __restrict__ idx,
                                 _Float16* __restrict__ members_h,
                                 _Float16* __restrict__ membersT_h) {
  int t = blockIdx.x * blockDim.x + threadIdx.x;
  int m = t >> 7, f = t & 127;
  if (m < MDIM) {
    _Float16 v = (_Float16)user_emb[(size_t)idx[m] * FDIM + f];
    members_h[(size_t)m * FDIM + f]  = v;
    membersT_h[(size_t)f * MDIM + m] = v;
  }
}

// COO SpMM scatter: Y[rows[e],:] += vals[e] * X[cols[e],:]; 32 lanes x float4 per edge
__global__ void k_spmm(const int* __restrict__ rows, const int* __restrict__ cols,
                       const float* __restrict__ vals, const float* __restrict__ X,
                       float* __restrict__ Y, int nnz) {
  long long t = (long long)blockIdx.x * blockDim.x + threadIdx.x;
  int e = (int)(t >> 5), f = (int)((t & 31) << 2);
  if (e < nnz) {
    const float4 x = *(const float4*)(X + (size_t)cols[e] * FDIM + f);
    const float v = vals[e];
    float* y = Y + (size_t)rows[e] * FDIM + f;
    atomicAdd(y + 0, v * x.x);
    atomicAdd(y + 1, v * x.y);
    atomicAdd(y + 2, v * x.z);
    atomicAdd(y + 3, v * x.w);
  }
}

// ---------------------------------------------------------------------------
// Attention pass 1: colsum[m] = sum_i exp(dot(item_i, member_m))
// block = 8 waves, each wave owns a 16-member column tile (B hoisted in VGPRs)
// ---------------------------------------------------------------------------
__global__ void k_colsum(const _Float16* __restrict__ item_h,
                         const _Float16* __restrict__ members_h,
                         float* __restrict__ colsum) {
  const int lane = threadIdx.x & 31;
  const int wave = threadIdx.x >> 5;
  const int m0 = blockIdx.x * 128 + wave * 16;
  const int i_begin = blockIdx.y * (IDIM / ISPLIT);
  v16h bt[4];
#pragma unroll
  for (int kc = 0; kc < 4; ++kc) bt[kc] = load_b_f16(members_h, FDIM, m0, kc * 32, lane);
  float part = 0.f;
  for (int it = 0; it < (IDIM / ISPLIT) / 16; ++it) {
    const int i0 = i_begin + it * 16;
    v8f acc = {0.f, 0.f, 0.f, 0.f, 0.f, 0.f, 0.f, 0.f};
#pragma unroll
    for (int kc = 0; kc < 4; ++kc)
      acc = wmma16(load_a_f16(item_h, FDIM, i0, kc * 32, lane), bt[kc], acc);
#pragma unroll
    for (int r = 0; r < 8; ++r) part += __expf(acc[r]);
  }
  part += __shfl_xor(part, 16, 32);  // combine lane-halves (same column)
  if (lane < 16) atomicAdd(&colsum[m0 + lane], part);
}

// ---------------------------------------------------------------------------
// Attention pass 2 (flash-style): att_item[i,f] = item[i,f] * sum_m w[i,m]*members[m,f]
// 8 waves share one 16-item stripe; item A-tiles hoisted out of the member loop.
// ---------------------------------------------------------------------------
__global__ void k_attention(const _Float16* __restrict__ item_h,
                            const _Float16* __restrict__ members_h,
                            const _Float16* __restrict__ membersT_h,
                            const float* __restrict__ colsum,
                            const float* __restrict__ item_emb,
                            float* __restrict__ att_item) {
  __shared__ __align__(32) _Float16 E[16 * 128];
  const int lane = threadIdx.x & 31;
  const int wave = threadIdx.x >> 5;
  const int i0   = blockIdx.x * 16;
  const int half = lane >> 4;
  const int n    = lane & 15;
  v16h ai[4];
#pragma unroll
  for (int kc = 0; kc < 4; ++kc) ai[kc] = load_a_f16(item_h, FDIM, i0, kc * 32, lane);
  v8f acc = {0.f, 0.f, 0.f, 0.f, 0.f, 0.f, 0.f, 0.f};
  for (int mc = 0; mc < MDIM; mc += 128) {
    const int mt = mc + wave * 16;
    if (mc + 128 < MDIM)  // cover HBM latency on the streamed B operand
      __builtin_prefetch(membersT_h + (size_t)(wave * 16 + n) * MDIM + mc + 128, 0, 1);
    v8f e = {0.f, 0.f, 0.f, 0.f, 0.f, 0.f, 0.f, 0.f};
#pragma unroll
    for (int kc = 0; kc < 4; ++kc)
      e = wmma16(ai[kc], load_b_f16(members_h, FDIM, mt, kc * 32, lane), e);
    const float inv = 1.f / colsum[mt + n];
#pragma unroll
    for (int r = 0; r < 8; ++r)
      E[(r + 8 * half) * 128 + wave * 16 + n] = (_Float16)(__expf(e[r]) * inv);
    __syncthreads();
#pragma unroll
    for (int kc = 0; kc < 4; ++kc)
      acc = wmma16(load_a_lds(E, 128, kc * 32, lane),
                   load_b_f16(membersT_h, MDIM, wave * 16, mc + kc * 32, lane), acc);
    __syncthreads();
  }
#pragma unroll
  for (int r = 0; r < 8; ++r) {
    const int row = i0 + r + 8 * half;
    const int col = wave * 16 + n;
    att_item[(size_t)row * FDIM + col] = acc[r] * item_emb[(size_t)row * FDIM + col];
  }
}

// ---------------------------------------------------------------------------
// Fused 5-way linear (+bias) + LeakyReLU + row L2-normalize.
// block = 8 waves over the 8 output 16-col tiles of one 16-row stripe.
// y = sum_k (X_k [* M_k]) @ W_k^T + b_k  (torch convention); W pre-converted f16
// ---------------------------------------------------------------------------
__global__ void k_lin5(const float* X0, const float* M0,
                       const float* X1, const float* M1,
                       const float* X2, const float* M2,
                       const float* X3, const float* M3,
                       const float* X4, const float* M4,
                       const _Float16* __restrict__ Wh,  // [5,F,F] row-major f16
                       const float* __restrict__ bias,   // [5,F]
                       float* __restrict__ out) {
  __shared__ float rowsq[16];
  const int lane = threadIdx.x & 31;
  const int wave = threadIdx.x >> 5;
  const int i0   = blockIdx.x * 16;
  const int j0   = wave * 16;
  const int half = lane >> 4;
  const int n    = lane & 15;
  if (threadIdx.x < 16) rowsq[threadIdx.x] = 0.f;
  __syncthreads();

  const float* Xs[5] = {X0, X1, X2, X3, X4};
  const float* Ms[5] = {M0, M1, M2, M3, M4};
  v8f acc = {0.f, 0.f, 0.f, 0.f, 0.f, 0.f, 0.f, 0.f};
#pragma unroll
  for (int k = 0; k < 5; ++k) {
    const _Float16* Wk = Wh + (size_t)k * FDIM * FDIM;
#pragma unroll
    for (int kc = 0; kc < 4; ++kc)
      acc = wmma16(load_a_f32(Xs[k], FDIM, i0, kc * 32, Ms[k], lane),
                   load_b_f16(Wk, FDIM, j0, kc * 32, lane), acc);
  }
  float bsum = 0.f;
#pragma unroll
  for (int k = 0; k < 5; ++k) bsum += bias[k * FDIM + j0 + n];

  float vr[8];
#pragma unroll
  for (int r = 0; r < 8; ++r) {
    float v = acc[r] + bsum;
    v = (v >= 0.f) ? v : NEG_SLOPE * v;  // LeakyReLU
    vr[r] = v;
    atomicAdd(&rowsq[r + 8 * half], v * v);
  }
  __syncthreads();
#pragma unroll
  for (int r = 0; r < 8; ++r) {
    const int row = r + 8 * half;
    const float scale = 1.f / fmaxf(sqrtf(rowsq[row]), 1e-12f);
    out[(size_t)(i0 + row) * FDIM + j0 + n] = vr[r] * scale;
  }
}

// ---------------------------------------------------------------------------
extern "C" void kernel_launch(void* const* d_in, const int* in_sizes, int n_in,
                              void* d_out, int out_size, void* d_ws, size_t ws_size,
                              hipStream_t stream) {
  const float* group_emb = (const float*)d_in[0];
  const float* user_emb  = (const float*)d_in[1];
  const float* item_emb  = (const float*)d_in[2];
  const int*   member_idx = (const int*)d_in[3];
  const int*   rui_rows = (const int*)d_in[4];
  const int*   rui_cols = (const int*)d_in[5];
  const float* rui_vals = (const float*)d_in[6];
  const int*   rgu_rows = (const int*)d_in[7];
  const int*   rgu_cols = (const int*)d_in[8];
  const float* rgu_vals = (const float*)d_in[9];
  const int*   rgi_rows = (const int*)d_in[10];
  const int*   rgi_cols = (const int*)d_in[11];
  const float* rgi_vals = (const float*)d_in[12];
  const float* Wu = (const float*)d_in[13];
  const float* bu = (const float*)d_in[14];
  const float* Wi = (const float*)d_in[15];
  const float* bi = (const float*)d_in[16];
  const float* Wg = (const float*)d_in[17];
  const float* bg = (const float*)d_in[18];
  const int nnz_rui = in_sizes[4];
  const int nnz_rgu = in_sizes[7];
  const int nnz_rgi = in_sizes[10];

  // workspace layout (64B-aligned sub-allocations)
  char* ws = (char*)d_ws;
  size_t off = 0;
  auto allocF = [&](size_t n) { float* p = (float*)(ws + off);
                                off = (off + n * 4 + 63) & ~(size_t)63; return p; };
  auto allocH = [&](size_t n) { _Float16* p = (_Float16*)(ws + off);
                                off = (off + n * 2 + 63) & ~(size_t)63; return p; };
  _Float16* item_h     = allocH((size_t)IDIM * FDIM);
  _Float16* members_h  = allocH((size_t)MDIM * FDIM);
  _Float16* membersT_h = allocH((size_t)FDIM * MDIM);
  _Float16* wg_h       = allocH((size_t)5 * FDIM * FDIM);
  _Float16* wu_h       = allocH((size_t)5 * FDIM * FDIM);
  _Float16* wi_h       = allocH((size_t)5 * FDIM * FDIM);
  float* att_item = allocF((size_t)IDIM * FDIM);
  float* zbase    = (float*)(ws + off);     // accumulators needing zero-init
  float* colsum   = allocF(MDIM);
  float* rui_ei   = allocF((size_t)UDIM * FDIM);
  float* rgu_t_eg = allocF((size_t)UDIM * FDIM);
  float* rgi_t_eg = allocF((size_t)IDIM * FDIM);
  float* rui_t_eu = allocF((size_t)IDIM * FDIM);
  float* rgi_ei   = allocF((size_t)GDIM * FDIM);
  float* rgu_eu   = allocF((size_t)GDIM * FDIM);
  float* att_g    = allocF((size_t)GDIM * FDIM);
  const long long nz = (long long)(((float*)(ws + off)) - zbase);

  float* out_group = (float*)d_out;
  float* out_user  = out_group + (size_t)GDIM * FDIM;
  float* out_item  = out_user + (size_t)UDIM * FDIM;

  // 1) zero accumulators
  k_zero<<<(int)((nz + 255) / 256), 256, 0, stream>>>(zbase, nz);

  // 2) f16 copies of heavily-reused operands (L2-resident afterwards)
  k_cvt_f16<<<(IDIM * FDIM) / 256, 256, 0, stream>>>(item_emb, item_h, IDIM * FDIM);
  k_cvt_f16<<<(5 * FDIM * FDIM + 255) / 256, 256, 0, stream>>>(Wg, wg_h, 5 * FDIM * FDIM);
  k_cvt_f16<<<(5 * FDIM * FDIM + 255) / 256, 256, 0, stream>>>(Wu, wu_h, 5 * FDIM * FDIM);
  k_cvt_f16<<<(5 * FDIM * FDIM + 255) / 256, 256, 0, stream>>>(Wi, wi_h, 5 * FDIM * FDIM);

  // 3) gather members (+ transpose), f16
  k_gather_members<<<(MDIM * FDIM) / 256, 256, 0, stream>>>(user_emb, member_idx,
                                                            members_h, membersT_h);

  // 4) attention softmax column sums
  k_colsum<<<dim3(MDIM / 128, ISPLIT), 256, 0, stream>>>(item_h, members_h, colsum);

  // 5) attention pass 2 -> attentive_item
  k_attention<<<IDIM / 16, 256, 0, stream>>>(item_h, members_h, membersT_h, colsum,
                                             item_emb, att_item);

  // 6) SpMMs (atomic scatter)
  auto spmm_grid = [](int nnz) { return (int)(((long long)nnz * 32 + 255) / 256); };
  k_spmm<<<spmm_grid(nnz_rui), 256, 0, stream>>>(rui_rows, rui_cols, rui_vals, item_emb, rui_ei,   nnz_rui);
  k_spmm<<<spmm_grid(nnz_rgu), 256, 0, stream>>>(rgu_cols, rgu_rows, rgu_vals, group_emb, rgu_t_eg, nnz_rgu);
  k_spmm<<<spmm_grid(nnz_rgi), 256, 0, stream>>>(rgi_cols, rgi_rows, rgi_vals, group_emb, rgi_t_eg, nnz_rgi);
  k_spmm<<<spmm_grid(nnz_rui), 256, 0, stream>>>(rui_cols, rui_rows, rui_vals, user_emb, rui_t_eu, nnz_rui);
  k_spmm<<<spmm_grid(nnz_rgi), 256, 0, stream>>>(rgi_rows, rgi_cols, rgi_vals, item_emb, rgi_ei,   nnz_rgi);
  k_spmm<<<spmm_grid(nnz_rgu), 256, 0, stream>>>(rgu_rows, rgu_cols, rgu_vals, user_emb, rgu_eu,   nnz_rgu);
  k_spmm<<<spmm_grid(nnz_rgi), 256, 0, stream>>>(rgi_rows, rgi_cols, rgi_vals, att_item, att_g,    nnz_rgi);

  // 7) fused lin5 + LeakyReLU + L2-normalize, straight into d_out
  // group: [g, rgi_ei, rgu_eu*g, rgi_ei*g, att_g]
  k_lin5<<<GDIM / 16, 256, 0, stream>>>(group_emb, nullptr, rgi_ei, nullptr,
                                        rgu_eu, group_emb, rgi_ei, group_emb,
                                        att_g, nullptr, wg_h, bg, out_group);
  // user: [u, rui_ei, rui_ei*u, rgu_t_eg*u, rgu_t_eg]
  k_lin5<<<UDIM / 16, 256, 0, stream>>>(user_emb, nullptr, rui_ei, nullptr,
                                        rui_ei, user_emb, rgu_t_eg, user_emb,
                                        rgu_t_eg, nullptr, wu_h, bu, out_user);
  // item: [i, rui_t_eu, rui_t_eu*i, rgi_t_eg*i, rgi_t_eg]
  k_lin5<<<IDIM / 16, 256, 0, stream>>>(item_emb, nullptr, rui_t_eu, nullptr,
                                        rui_t_eu, item_emb, rgi_t_eg, item_emb,
                                        rgi_t_eg, nullptr, wi_h, bi, out_item);
}
